// PointNet2Layer_85315230368306
// MI455X (gfx1250) — matrix operand order repaired and checked by reading, not compiled, not run on hardware
//
#include <hip/hip_runtime.h>

// ---------------- problem constants ----------------
constexpr int B = 8, P = 8192, F = 32, M = 2048, K = 64;
constexpr float R2 = 0.04f;   // R = 0.2

typedef __attribute__((ext_vector_type(16))) __bf16 v16bf;
typedef __attribute__((ext_vector_type(8)))  float  v8f;

struct Frag {
    union {
        v16bf bf;
        unsigned int u[8];
    };
};

__device__ __forceinline__ unsigned short f2bf(float f) {
    unsigned int u = __float_as_uint(f);
    unsigned int r = u + 0x7FFFu + ((u >> 16) & 1u);   // round-to-nearest-even
    return (unsigned short)(r >> 16);
}
__device__ __forceinline__ unsigned int pack2(unsigned short lo, unsigned short hi) {
    return (unsigned int)lo | ((unsigned int)hi << 16);
}

// ---- A fragment: 16x32 A operand in this code's contiguous k order (same as load_b), source = LDS row-major [64][64] bf16
// lane L (row m = mt*16 + L%16): regs i hold K = kt*32 + 16*(L>=16) + 2i, 2i+1 (contiguous)
__device__ __forceinline__ Frag load_a(const unsigned int* S, int mt, int kt, int lane) {
    Frag f;
    int lhalf = lane >> 4, lrow = lane & 15;
    const unsigned int* rp = S + (mt * 16 + lrow) * 32 + kt * 16 + lhalf * 8;
#pragma unroll
    for (int i = 0; i < 4; ++i) f.u[i] = rp[i];
#pragma unroll
    for (int i = 0; i < 4; ++i) f.u[4 + i] = rp[4 + i];
    return f;
}

// ---- B fragment: 32x16 B operand in this code's contiguous k order (same as load_a), source = global Wt [N][64] bf16 (transposed weights)
// lane L (col n = nt*16 + L%16): regs i hold K = kt*32 + 16*(L>=16) + 2i, 2i+1 (contiguous!)
__device__ __forceinline__ Frag load_b(const unsigned short* Wt, int nt, int kt, int lane) {
    Frag f;
    int lhalf = lane >> 4, lrow = lane & 15;
    const unsigned int* wp =
        (const unsigned int*)(Wt + (nt * 16 + lrow) * 64 + kt * 32 + lhalf * 16);
#pragma unroll
    for (int i = 0; i < 8; ++i) f.u[i] = wp[i];
    return f;
}

// ---- 64x64 = 64x64 @ 64x64 GEMM, bias+ReLU epilogue, bf16 result to LDS (row-major [64][64])
__device__ __forceinline__ void gemm64_relu(const unsigned int* Asrc, const unsigned short* Bt,
                                            const float* bias, unsigned int* Dst, int lane) {
    int lhalf = lane >> 4, lrow = lane & 15;
    unsigned short* D16 = (unsigned short*)Dst;
#pragma unroll
    for (int nt = 0; nt < 4; ++nt) {
        int n = nt * 16 + lrow;
        float bv = bias[n];
#pragma unroll
        for (int mt = 0; mt < 4; ++mt) {
            v8f acc = {};
#pragma unroll
            for (int kt = 0; kt < 2; ++kt) {
                Frag a = load_a(Asrc, mt, kt, lane);
                Frag b = load_b(Bt, nt, kt, lane);
                acc = __builtin_amdgcn_wmma_f32_16x16x32_bf16(
                    false, a.bf, false, b.bf, (short)0, acc, false, false);
            }
#pragma unroll
            for (int r = 0; r < 8; ++r) {
                int m = mt * 16 + r + lhalf * 8;                   // D layout: M = r + 8*lhalf
                float v = fmaxf(acc[r] + bv, 0.0f);
                D16[m * 64 + n] = f2bf(v);
            }
        }
    }
}

// ================= kernel 0: transpose/pad/convert weights to bf16 =================
__global__ __launch_bounds__(256) void prep_kernel(const float* W1, const float* W2,
                                                   const float* W3, unsigned short* W1t,
                                                   unsigned short* W2t, unsigned short* W3t) {
    int t = blockIdx.x * 256 + threadIdx.x;
    if (t < 4096) {                       // W1: [35][64] -> W1t[n][k], pad k to 64
        int n = t >> 6, k = t & 63;
        W1t[n * 64 + k] = (k < 35) ? f2bf(W1[k * 64 + n]) : (unsigned short)0;
    } else if (t < 8192) {                // W2: [64][64] -> W2t[n][k]
        int e = t - 4096; int n = e >> 6, k = e & 63;
        W2t[n * 64 + k] = f2bf(W2[k * 64 + n]);
    } else if (t < 16384) {               // W3: [64][128] -> W3t[n][k]
        int e = t - 8192; int n = e >> 6, k = e & 63;
        W3t[n * 64 + k] = f2bf(W3[k * 128 + n]);
    }
}

// ================= kernel 1: farthest point sampling (1 block per cloud) =================
__global__ __launch_bounds__(1024) void fps_kernel(const float* __restrict__ pos,
                                                   float* __restrict__ cent_ws,
                                                   float* __restrict__ cent_out) {
    const int b = blockIdx.x;
    const float* pb = pos + (size_t)b * P * 3;
    __shared__ float dist[P];          // 32 KB running min-distances
    __shared__ float sv[32];
    __shared__ int   si[32];
    __shared__ float lp[3];
    __shared__ int   lasts;

    int tid = threadIdx.x;
    int lane = tid & 31, wave = tid >> 5;
    for (int i = tid; i < P; i += 1024) dist[i] = 3.0e38f;
    if (tid == 0) {
        lasts = 0;
        float px = pb[0], py = pb[1], pz = pb[2];
        float* c0 = cent_out + (size_t)(b * M) * 3;
        float* c1 = cent_ws + (size_t)(b * M) * 3;
        c0[0] = px; c0[1] = py; c0[2] = pz;
        c1[0] = px; c1[1] = py; c1[2] = pz;
    }
    __syncthreads();

    for (int m = 1; m < M; ++m) {
        if (tid == 0) {
            int l = lasts;
            lp[0] = pb[l * 3]; lp[1] = pb[l * 3 + 1]; lp[2] = pb[l * 3 + 2];
        }
        __syncthreads();
        float bx = lp[0], by = lp[1], bz = lp[2];
        float bestv = -1.0f; int besti = 0;
        for (int i = tid; i < P; i += 1024) {
            float dx = pb[i * 3] - bx, dy = pb[i * 3 + 1] - by, dz = pb[i * 3 + 2] - bz;
            float d = dx * dx + dy * dy + dz * dz;
            float nd = fminf(dist[i], d);
            dist[i] = nd;
            if (nd > bestv) { bestv = nd; besti = i; }
        }
        // wave32 shuffle reduce (prefer lower index on ties)
#pragma unroll
        for (int s = 16; s > 0; s >>= 1) {
            float ov = __shfl_xor(bestv, s, 32);
            int   oi = __shfl_xor(besti, s, 32);
            if (ov > bestv || (ov == bestv && oi < besti)) { bestv = ov; besti = oi; }
        }
        if (lane == 0) { sv[wave] = bestv; si[wave] = besti; }
        __syncthreads();
        if (wave == 0) {
            bestv = sv[lane]; besti = si[lane];
#pragma unroll
            for (int s = 16; s > 0; s >>= 1) {
                float ov = __shfl_xor(bestv, s, 32);
                int   oi = __shfl_xor(besti, s, 32);
                if (ov > bestv || (ov == bestv && oi < besti)) { bestv = ov; besti = oi; }
            }
            if (lane == 0) {
                lasts = besti;
                float px = pb[besti * 3], py = pb[besti * 3 + 1], pz = pb[besti * 3 + 2];
                float* c0 = cent_out + (size_t)(b * M + m) * 3;
                float* c1 = cent_ws + (size_t)(b * M + m) * 3;
                c0[0] = px; c0[1] = py; c0[2] = pz;
                c1[0] = px; c1[1] = py; c1[2] = pz;
            }
        }
        __syncthreads();
    }
}

// ================= kernel 2: ball query (1 wave per centroid) =================
__global__ __launch_bounds__(256) void ball_kernel(const float* __restrict__ pos,
                                                   const float* __restrict__ cent_ws,
                                                   int* __restrict__ nbo, int* __restrict__ cnts,
                                                   float* __restrict__ batch_out) {
    int gt = blockIdx.x * 256 + threadIdx.x;
    int wid = gt >> 5;                 // centroid id, exact grid -> wid < B*M
    int lane = gt & 31;
    int b = wid / M;
    const float* pb = pos + (size_t)b * P * 3;
    float cx = cent_ws[wid * 3], cy = cent_ws[wid * 3 + 1], cz = cent_ws[wid * 3 + 2];
    int* nbw = nbo + (size_t)wid * K;

    int cnt = 0;
    for (int base = 0; base < P && cnt < K; base += 32) {
        int i = base + lane;
        float dx = pb[i * 3] - cx, dy = pb[i * 3 + 1] - cy, dz = pb[i * 3 + 2] - cz;
        bool in = (dx * dx + dy * dy + dz * dz) <= R2;
        unsigned mask = (unsigned)__ballot(in);
        if (in) {
            int r = __popc(mask & ((1u << lane) - 1u));
            int p = cnt + r;
            if (p < K) nbw[p] = i;     // lowest indices first, chunk order = index order
        }
        cnt += __popc(mask);
    }
    if (cnt > K) cnt = K;
    if (lane == 0) {
        cnts[wid] = cnt;
        batch_out[wid] = (float)(wid / M);
    }
}

// ================= kernel 3: gather -> bf16 WMMA MLP -> masked max (1 wave / centroid) ======
constexpr int WPB = 2;   // waves per block
__global__ __launch_bounds__(WPB * 32) void mlp_kernel(
    const float* __restrict__ pos, const float* __restrict__ x, const float* __restrict__ b1,
    const float* __restrict__ b2, const float* __restrict__ b3,
    const unsigned short* __restrict__ W1t, const unsigned short* __restrict__ W2t,
    const unsigned short* __restrict__ W3t, const float* __restrict__ cent_ws,
    const int* __restrict__ nbo, const int* __restrict__ cnts, float* __restrict__ out) {
    __shared__ unsigned int bufA[WPB][64 * 32];   // 8 KB / wave : msg, later h2
    __shared__ unsigned int bufB[WPB][64 * 32];   // 8 KB / wave : h1

    int w = threadIdx.x >> 5;
    int lane = threadIdx.x & 31;
    int c = blockIdx.x * WPB + w;                 // centroid id (grid exact)
    int cnt = cnts[c];
    int bcl = c / M;
    const float* pb = pos + (size_t)bcl * P * 3;
    const float* xb = x + (size_t)bcl * P * F;
    float cx = cent_ws[c * 3], cy = cent_ws[c * 3 + 1], cz = cent_ws[c * 3 + 2];
    const int* nbw = nbo + (size_t)c * K;
    unsigned int* A = bufA[w];
    unsigned int* Bf = bufB[w];

    // ---- build msg [64 rows][64 cols] bf16: [x_j (32) | pos_j - cent (3) | zero pad] ----
#pragma unroll
    for (int rr = 0; rr < 2; ++rr) {
        int j = lane + rr * 32;
        unsigned int* row = A + j * 32;
        if (j < cnt) {
            int id = nbw[j];
            const float* gx = xb + (size_t)id * F;
#pragma unroll
            for (int t = 0; t < 16; ++t)
                row[t] = pack2(f2bf(gx[2 * t]), f2bf(gx[2 * t + 1]));
            float d0 = pb[id * 3] - cx, d1 = pb[id * 3 + 1] - cy, d2v = pb[id * 3 + 2] - cz;
            row[16] = pack2(f2bf(d0), f2bf(d1));
            row[17] = pack2(f2bf(d2v), 0);
#pragma unroll
            for (int t = 18; t < 32; ++t) row[t] = 0;
        } else {
#pragma unroll
            for (int t = 0; t < 32; ++t) row[t] = 0;
        }
    }
    __syncthreads();

    gemm64_relu(A, W1t, b1, Bf, lane);    // h1 = relu(msg @ W1 + b1)
    __syncthreads();
    gemm64_relu(Bf, W2t, b2, A, lane);    // h2 = relu(h1  @ W2 + b2)
    __syncthreads();

    // ---- layer 3 (64x128) + masked max over valid neighbor rows, fused in D registers ----
    int lhalf = lane >> 4, lrow = lane & 15;
#pragma unroll
    for (int nt = 0; nt < 8; ++nt) {
        int n = nt * 16 + lrow;
        float bias = b3[n];
        float maxv = -3.4e38f;
#pragma unroll
        for (int mt = 0; mt < 4; ++mt) {
            v8f acc = {};
#pragma unroll
            for (int kt = 0; kt < 2; ++kt) {
                Frag a = load_a(A, mt, kt, lane);
                Frag b = load_b(W3t, nt, kt, lane);
                acc = __builtin_amdgcn_wmma_f32_16x16x32_bf16(
                    false, a.bf, false, b.bf, (short)0, acc, false, false);
            }
#pragma unroll
            for (int r = 0; r < 8; ++r) {
                int j = mt * 16 + r + lhalf * 8;
                if (j < cnt) {
                    float v = fmaxf(acc[r] + bias, 0.0f);
                    if (v > maxv) maxv = v;
                }
            }
        }
        float o = fmaxf(maxv, __shfl_xor(maxv, 16, 32));  // fold M-halves across lane halves
        if (lane < 16) out[(size_t)c * 128 + n] = o;
    }
}

// ===================================================================================
extern "C" void kernel_launch(void* const* d_in, const int* in_sizes, int n_in,
                              void* d_out, int out_size, void* d_ws, size_t ws_size,
                              hipStream_t stream) {
    const float* pos = (const float*)d_in[0];
    // d_in[1] (batch) is implied by layout; unused
    const float* x  = (const float*)d_in[2];
    const float* W1 = (const float*)d_in[3];
    const float* b1 = (const float*)d_in[4];
    const float* W2 = (const float*)d_in[5];
    const float* b2 = (const float*)d_in[6];
    const float* W3 = (const float*)d_in[7];
    const float* b3 = (const float*)d_in[8];

    float* out       = (float*)d_out;
    float* cent_out  = out;                                   // [B*M*3]
    float* feat_out  = out + (size_t)B * M * 3;               // [B*M*128]
    float* batch_out = feat_out + (size_t)B * M * 128;        // [B*M]

    char* ws = (char*)d_ws;
    unsigned short* W1t = (unsigned short*)(ws);              //  8 KB
    unsigned short* W2t = (unsigned short*)(ws + 8192);       //  8 KB
    unsigned short* W3t = (unsigned short*)(ws + 16384);      // 16 KB
    int*   cnts    = (int*)(ws + 32768);                      // 64 KB
    int*   nbb     = (int*)(ws + 98304);                      //  4 MB
    float* cent_ws = (float*)(ws + 98304 + (size_t)B * M * K * 4);

    prep_kernel<<<64, 256, 0, stream>>>(W1, W2, W3, W1t, W2t, W3t);
    fps_kernel<<<B, 1024, 0, stream>>>(pos, cent_ws, cent_out);
    ball_kernel<<<(B * M) / 8, 256, 0, stream>>>(pos, cent_ws, nbb, cnts, batch_out);
    mlp_kernel<<<(B * M) / WPB, WPB * 32, 0, stream>>>(pos, x, b1, b2, b3, W1t, W2t, W3t,
                                                       cent_ws, nbb, cnts, feat_out);
}
